// MyRNN_26688926777389
// MI455X (gfx1250) — matrix operand order repaired and checked by reading, not compiled, not run on hardware
//
#include <hip/hip_runtime.h>
#include <hip/hip_bf16.h>
#include <math.h>

// ---------------------------------------------------------------------------
// Stacked SimpleRNN (2 layers, tanh) persistent-state WMMA kernel for gfx1250.
//   B=1024, T=128, E=128, U=512.
//   - bf16 WMMA (v_wmma_f32_16x16x32_bf16), f32 accumulate
//   - per-batch-row recurrence => rows partitioned across workgroups,
//     h0/h1 live in LDS for all 128 timesteps (no global sync needed)
//   - weights pre-packed into WMMA B-fragment order => global b128 loads
//     with compile-time immediate offsets
//   - copy-free ping-pong software pipeline (loads for kk+1 in flight over
//     the WMMAs of kk)
//   - opaque-zero SGPR offset blocks LICM of the Wx0 fragments
//   - waves_per_eu(2,2): matches the structural occupancy (one 8-wave WG per
//     WGP), freeing the scheduler to hold fragments live across WMMA groups
// ---------------------------------------------------------------------------

typedef unsigned int u32;
typedef __attribute__((ext_vector_type(4)))  u32    u32x4;
typedef __attribute__((ext_vector_type(8)))  float  v8f;
typedef __attribute__((ext_vector_type(16))) __bf16 v16bf;

#define B_DIM 1024
#define T_DIM 128
#define E_DIM 128
#define U_DIM 512
#define ROWS  16            // batch rows per workgroup -> 64 workgroups
#define NT_U  (U_DIM/16)    // 32 N-tiles across U
#define KT_U  (U_DIM/32)    // 16 K-steps for U-dim reduction
#define KT_E  (E_DIM/32)    // 4  K-steps for E-dim reduction

union Frag { u32x4 u[2]; v16bf v; };

// Native transcendental activation (gfx1250 has V_TANH_F32 as a TRANS op).
__device__ __forceinline__ float act_tanh(float x) {
#if __has_builtin(__builtin_amdgcn_tanhf)
    return __builtin_amdgcn_tanhf(x);
#else
    float e = __builtin_amdgcn_exp2f(x * 2.885390081777927f);
    return 1.0f - 2.0f * __builtin_amdgcn_rcpf(e + 1.0f);
#endif
}

__device__ __forceinline__ float fast_sigmoid(float x) {
    float e = __builtin_amdgcn_exp2f(-x * 1.4426950408889634f);
    return __builtin_amdgcn_rcpf(1.0f + e);
}

// A fragment from per-lane base (row-major LDS) + constant element offset.
// ISA A-layout: lanes 0-15 row M=lane, VGPR0..3 = K 0..7, VGPR4..7 = K 16..23;
// lanes 16-31 same rows, K offsets +8.  => two contiguous 16B LDS loads.
__device__ __forceinline__ v16bf load_a_at(const __bf16* lanebase, int elemOff) {
    const u32x4* p = (const u32x4*)(lanebase + elemOff);
    Frag f;
    f.u[0] = p[0];   // K +0 .. +7
    f.u[1] = p[2];   // K +16 .. +23
    return f.v;
}

// B fragment from per-lane base (pre-packed weights) + constant element offset.
__device__ __forceinline__ v16bf load_b_at(const __bf16* lanebase, int elemOff) {
    const u32x4* p = (const u32x4*)(lanebase + elemOff);
    Frag f;
    f.u[0] = p[0];
    f.u[1] = p[1];
    return f.v;
}

// Copy-free ping-pong pipelined GEMM: acc[i] += A(abuf) x W(nt-tiles nt0..nt0+3).
// Fully unrolled; fragment offsets fold into instruction immediates.
template <int KT, int LDK>
__device__ __forceinline__ void gemm_pipe(v8f (&acc)[4], const __bf16* abuf,
                                          const __bf16* __restrict__ wp,
                                          int nt0, int lane) {
    const __bf16* albase = abuf + (lane & 15) * LDK + ((lane >> 4) << 3);
    const __bf16* wlbase = wp + (size_t)nt0 * 512 + lane * 16;

    v16bf a[2];
    v16bf b[2][4];
    a[0] = load_a_at(albase, 0);
#pragma unroll
    for (int i = 0; i < 4; ++i) b[0][i] = load_b_at(wlbase, i * 512);

#pragma unroll
    for (int kk = 0; kk < KT; ++kk) {
        const int cur = kk & 1, nxt = cur ^ 1;
        if (kk + 1 < KT) {
            a[nxt] = load_a_at(albase, (kk + 1) * 32);
#pragma unroll
            for (int i = 0; i < 4; ++i)
                b[nxt][i] = load_b_at(wlbase, ((kk + 1) * NT_U + i) * 512);
        }
#pragma unroll
        for (int i = 0; i < 4; ++i)
            acc[i] = __builtin_amdgcn_wmma_f32_16x16x32_bf16(
                         false, a[cur], false, b[cur][i], (short)0, acc[i], false, false);
    }
}

// ---------------------------------------------------------------------------
// Prep: gather embeddings -> bf16, time-major [T][B][E]
// ---------------------------------------------------------------------------
__global__ __launch_bounds__(256) void gather_embed_kernel(
    const int* __restrict__ tokens, const float* __restrict__ emb,
    __bf16* __restrict__ Xbf)
{
    const int idx = blockIdx.x * 256 + threadIdx.x;   // idx = t*B + b
    const int t = idx >> 10;
    const int b = idx & (B_DIM - 1);
    const int tok = tokens[b * T_DIM + t];
    const float4* src = (const float4*)(emb + (size_t)tok * E_DIM);
    __bf16* dst = Xbf + (size_t)idx * E_DIM;
#pragma unroll 8
    for (int e = 0; e < E_DIM / 4; ++e) {
        float4 f = src[e];
        dst[4*e+0] = (__bf16)f.x;  dst[4*e+1] = (__bf16)f.y;
        dst[4*e+2] = (__bf16)f.z;  dst[4*e+3] = (__bf16)f.w;
    }
}

// ---------------------------------------------------------------------------
// Prep: pack f32 weight [K][N] into bf16 WMMA B-fragment order.
// Per 32x16 tile (kt,nt): lane holds column n = nt*16 + (lane&15),
// K = kt*32 + (lane>=16 ? 8 : 0) + (e<8 ? e : 16 + e-8), 16 values (VGPR v pair 2v,2v+1).
// ---------------------------------------------------------------------------
__global__ __launch_bounds__(256) void pack_w_kernel(
    const float* __restrict__ src, __bf16* __restrict__ dst, int K, int N)
{
    const int tid = blockIdx.x * 256 + threadIdx.x;
    if (tid >= K * N) return;
    const int blk  = tid >> 9;          // 512 elements per tile
    const int idx  = tid & 511;
    const int lane = idx >> 4;
    const int e    = idx & 15;          // element within lane = v*2 + j
    const int NT   = N >> 4;
    const int kt   = blk / NT;
    const int nt   = blk - kt * NT;
    const int n    = nt * 16 + (lane & 15);
    const int k    = kt * 32 + ((lane >> 4) << 3) + (e & 7) + ((e >> 3) << 4);
    dst[tid] = (__bf16)src[(size_t)k * N + n];
}

// ---------------------------------------------------------------------------
// Main persistent RNN kernel: 64 WGs x 256 threads (8 waves).
// Wave w handles N-tiles [4w, 4w+4) for all 16 rows of its workgroup.
// ---------------------------------------------------------------------------
__global__ __launch_bounds__(256)
__attribute__((amdgpu_waves_per_eu(2, 2)))
void rnn_persist_kernel(
    const __bf16* __restrict__ Xbf,
    const __bf16* __restrict__ Wx0p, const __bf16* __restrict__ Wh0p,
    const __bf16* __restrict__ Wx1p, const __bf16* __restrict__ Wh1p,
    const float*  __restrict__ b0,   const float*  __restrict__ b1,
    const float*  __restrict__ Wo,   const float*  __restrict__ bo,
    float* __restrict__ out)
{
    __shared__ __bf16 xbuf [ROWS * E_DIM];   // 4 KB
    __shared__ __bf16 h0buf[ROWS * U_DIM];   // 16 KB
    __shared__ __bf16 h1buf[ROWS * U_DIM];   // 16 KB

    const int tid     = threadIdx.x;
    const int lane    = tid & 31;
    const int wave    = tid >> 5;            // 0..7
    const int rowBase = blockIdx.x * ROWS;
    const int nt0     = wave * 4;            // first of 4 N-tiles
    const int colBase = nt0 * 16;
    const int drow    = (lane >> 4) << 3;    // D-layout row offset (0 or 8)

    // zero-init recurrent state
    {
        u32* h0w = (u32*)h0buf; u32* h1w = (u32*)h1buf;
        for (int i = tid; i < ROWS * U_DIM / 2; i += 256) { h0w[i] = 0u; h1w[i] = 0u; }
    }
    __syncthreads();

    float bias0[4], bias1[4];
#pragma unroll
    for (int i = 0; i < 4; ++i) {
        bias0[i] = b0[colBase + i * 16 + (lane & 15)];
        bias1[i] = b1[colBase + i * 16 + (lane & 15)];
    }

    const v8f vzero = {0.f, 0.f, 0.f, 0.f, 0.f, 0.f, 0.f, 0.f};

    // register double-buffer for the x tile: load t=0 now, t+1 during step t
    const u32x4* xsrc = (const u32x4*)(Xbf + (size_t)rowBase * E_DIM);
    const size_t xstep = (size_t)B_DIM * E_DIM / 8;   // u32x4 elements per t
    u32x4 xreg = xsrc[tid];

    for (int t = 0; t < T_DIM; ++t) {
        // ---- commit staged x_t tile (16 rows x 128) to LDS ----------------
        ((u32x4*)xbuf)[tid] = xreg;
        __syncthreads();
        if (t + 1 < T_DIM)                  // overlap next-step load w/ compute
            xreg = xsrc[(t + 1) * xstep + tid];

        // Opaque zero (fresh SGPR each iteration): blocks LICM of the Wx0
        // fragment loads without losing the global address space.
        u32 zoff;
        asm volatile("s_mov_b32 %0, 0" : "=s"(zoff));

        // ================= layer 0: h0' = tanh(x@Wx0 + h0@Wh0 + b0) ========
        v8f acc[4];
#pragma unroll
        for (int i = 0; i < 4; ++i) acc[i] = vzero;

        gemm_pipe<KT_E, E_DIM>(acc, xbuf,  Wx0p + zoff, nt0, lane);   // x  @ Wx0
        gemm_pipe<KT_U, U_DIM>(acc, h0buf, Wh0p,        nt0, lane);   // h0 @ Wh0

        float hn[4][8];
#pragma unroll
        for (int i = 0; i < 4; ++i)
#pragma unroll
            for (int r = 0; r < 8; ++r)
                hn[i][r] = act_tanh(acc[i][r] + bias0[i]);

        __syncthreads();                    // all reads of old h0 done
#pragma unroll
        for (int i = 0; i < 4; ++i)
#pragma unroll
            for (int r = 0; r < 8; ++r)     // D layout: VGPR r -> row r+drow
                h0buf[(r + drow) * U_DIM + colBase + i * 16 + (lane & 15)] = (__bf16)hn[i][r];
        __syncthreads();

        // ================= layer 1: h1' = tanh(h0'@Wx1 + h1@Wh1 + b1) ======
#pragma unroll
        for (int i = 0; i < 4; ++i) acc[i] = vzero;

        gemm_pipe<KT_U, U_DIM>(acc, h0buf, Wx1p, nt0, lane);   // h0' @ Wx1
        gemm_pipe<KT_U, U_DIM>(acc, h1buf, Wh1p, nt0, lane);   // h1  @ Wh1

#pragma unroll
        for (int i = 0; i < 4; ++i)
#pragma unroll
            for (int r = 0; r < 8; ++r)
                hn[i][r] = act_tanh(acc[i][r] + bias1[i]);

        __syncthreads();                    // all reads of old h1 done
#pragma unroll
        for (int i = 0; i < 4; ++i)
#pragma unroll
            for (int r = 0; r < 8; ++r)
                h1buf[(r + drow) * U_DIM + colBase + i * 16 + (lane & 15)] = (__bf16)hn[i][r];
        __syncthreads();
    }

    // ---- head: sigmoid(h1 @ Wo + bo), one thread per row (tiny) ----------
    if (tid < ROWS) {
        float s = 0.f;
        for (int k = 0; k < U_DIM; ++k)
            s += (float)h1buf[tid * U_DIM + k] * Wo[k];
        s += bo[0];
        out[rowBase + tid] = fast_sigmoid(s);
    }
}

// ---------------------------------------------------------------------------
// Workspace layout (bytes):
//   Xbf  : T*B*E*2           = 33,554,432
//   Wx0p : E*U*2             =    131,072
//   Wh0p : U*U*2             =    524,288
//   Wx1p : U*U*2             =    524,288
//   Wh1p : U*U*2             =    524,288
//   total                    = 35,258,368
// ---------------------------------------------------------------------------
extern "C" void kernel_launch(void* const* d_in, const int* in_sizes, int n_in,
                              void* d_out, int out_size, void* d_ws, size_t ws_size,
                              hipStream_t stream) {
    const int*   tokens = (const int*)  d_in[0];
    const float* emb    = (const float*)d_in[1];
    const float* Wx0    = (const float*)d_in[2];
    const float* Wh0    = (const float*)d_in[3];
    const float* b0     = (const float*)d_in[4];
    const float* Wx1    = (const float*)d_in[5];
    const float* Wh1    = (const float*)d_in[6];
    const float* b1     = (const float*)d_in[7];
    const float* Wo     = (const float*)d_in[8];
    const float* bo     = (const float*)d_in[9];
    float* out = (float*)d_out;

    char* ws = (char*)d_ws;
    __bf16* Xbf  = (__bf16*)(ws);
    __bf16* Wx0p = (__bf16*)(ws + 33554432);
    __bf16* Wh0p = (__bf16*)(ws + 33554432 + 131072);
    __bf16* Wx1p = (__bf16*)(ws + 33554432 + 131072 + 524288);
    __bf16* Wh1p = (__bf16*)(ws + 33554432 + 131072 + 2 * 524288);

    // prep (re-run every call for determinism)
    gather_embed_kernel<<<(T_DIM * B_DIM) / 256, 256, 0, stream>>>(tokens, emb, Xbf);
    pack_w_kernel<<<(E_DIM * U_DIM) / 256, 256, 0, stream>>>(Wx0, Wx0p, E_DIM, U_DIM);
    pack_w_kernel<<<(U_DIM * U_DIM) / 256, 256, 0, stream>>>(Wh0, Wh0p, U_DIM, U_DIM);
    pack_w_kernel<<<(U_DIM * U_DIM) / 256, 256, 0, stream>>>(Wx1, Wx1p, U_DIM, U_DIM);
    pack_w_kernel<<<(U_DIM * U_DIM) / 256, 256, 0, stream>>>(Wh1, Wh1p, U_DIM, U_DIM);

    // persistent recurrence: 64 workgroups own 16 batch rows each
    rnn_persist_kernel<<<B_DIM / ROWS, 256, 0, stream>>>(
        Xbf, Wx0p, Wh0p, Wx1p, Wh1p, b0, b1, Wo, bo, out);
}
